// Attention_29686813950039
// MI455X (gfx1250) — matrix operand order repaired and checked by reading, not compiled, not run on hardware
//
#include <hip/hip_runtime.h>
#include <hip/hip_bf16.h>

// ---------------------------------------------------------------------------
// Gemma-style GQA attention block for gfx1250 (MI455X), bf16 WMMA pipeline.
//   B=2 T=2048 D=3584 NQ=16 NKV=8 H=256 WINDOW=1024 SCALE=0.0625
// Stages:
//   1) k_gemm_qkv : Y[4096][8192]bf16 = X * [Wq|Wk|Wv]       (wmma bf16)
//   2) k_norm_rope: RMSNorm + RoPE on q/k head vectors of Y  (f32 VALU)
//   3) k_attn     : sliding-window flash attention           (wmma bf16 + TDM)
//   4) k_gemm_out : Out[4096][3584]f32 = A * Wo              (wmma bf16)
// ---------------------------------------------------------------------------

typedef __attribute__((ext_vector_type(16))) __bf16 bf16x16;
typedef __attribute__((ext_vector_type(8)))  float  f32x8;
typedef unsigned int u32x4 __attribute__((ext_vector_type(4)));
typedef int          i32x8 __attribute__((ext_vector_type(8)));
typedef int          i32x4 __attribute__((ext_vector_type(4)));

#if defined(__gfx1250__) && __has_builtin(__builtin_amdgcn_tensor_load_to_lds) && \
    __has_builtin(__builtin_amdgcn_s_wait_tensorcnt)
#define USE_TDM 1
#else
#define USE_TDM 0
#endif

union FragU { bf16x16 v; unsigned int u[8]; };

// --- WMMA fragment loaders matching CDNA5 16-bit A/B VGPR layouts ----------
// A (16x32, MxK): lanes 0-15 row M=lane, k pairs {0,2,4,6,16,18,20,22};
//                 lanes 16-31 row M=lane-16, +8 on the k pairs.
__device__ __forceinline__ bf16x16 frag_a_lds(const __bf16* tile, int ld, int lane) {
  const int r  = lane & 15;
  const int kb = (lane >> 4) << 3;
  FragU f;
  const __bf16* row = tile + r * ld;
#pragma unroll
  for (int j = 0; j < 8; ++j) {
    const int k = kb + ((j < 4) ? (2 * j) : (16 + 2 * (j - 4)));
    f.u[j] = *(const unsigned int*)(row + k);
  }
  return f.v;
}

__device__ __forceinline__ bf16x16 frag_a_glb(const __bf16* tile, long ld, int lane) {
  const int r  = lane & 15;
  const int kb = (lane >> 4) << 3;
  FragU f;
  const __bf16* row = tile + (size_t)r * (size_t)ld;
#pragma unroll
  for (int j = 0; j < 8; ++j) {
    const int k = kb + ((j < 4) ? (2 * j) : (16 + 2 * (j - 4)));
    f.u[j] = *(const unsigned int*)(row + k);
  }
  return f.v;
}

// B (32x16, KxN) read from an N-major (transposed) LDS tile: lane n = lane&15,
// k pairs {0,2,4,6,16,18,20,22} for lanes 0-15, +8 for lanes 16-31 (same as A).
__device__ __forceinline__ bf16x16 frag_b_lds(const __bf16* tileT, int ld, int lane) {
  const int n  = lane & 15;
  const int kb = (lane >> 4) << 3;
  FragU f;
  const __bf16* row = tileT + n * ld;
#pragma unroll
  for (int j = 0; j < 8; ++j)
    f.u[j] = *(const unsigned int*)(row + kb + ((j < 4) ? (2 * j) : (16 + 2 * (j - 4))));
  return f.v;
}

__device__ __forceinline__ f32x8 wmma_bf16(bf16x16 a, bf16x16 b, f32x8 c) {
  return __builtin_amdgcn_wmma_f32_16x16x32_bf16(false, a, false, b, (short)0, c,
                                                 false, false);
}

// ---------------------------------------------------------------------------
// Kernel 1: fused QKV projection.  X[4096][3584]f32 x W -> Y[4096][8192]bf16
// Y columns: [0,4096) = q head (c>>8), [4096,6144) = k head, [6144,8192) = v.
// ---------------------------------------------------------------------------
__global__ __launch_bounds__(256) void k_gemm_qkv(const float* __restrict__ X,
                                                  const float* __restrict__ Wq,
                                                  const float* __restrict__ Wkv,
                                                  __bf16* __restrict__ Y) {
  __shared__ __bf16 sA[64 * 34];    // [row][k]
  __shared__ __bf16 sB[128 * 34];   // transposed: [n][k]
  const int tid  = threadIdx.x;
  const int lane = tid & 31;
  const int wave = tid >> 5;
  const int bm = blockIdx.x * 64;
  const int bn = blockIdx.y * 128;
  const int wm = (wave >> 2) * 32;
  const int wn = (wave & 3) * 32;

  f32x8 acc[2][2] = {};

  // per-thread B column group (16 consecutive cols, always within one head)
  const int krow = tid >> 3;
  const int nb   = (tid & 7) * 16;
  const int col0 = bn + nb;
  const float* wbase;
  if (col0 < 4096)
    wbase = Wq + (size_t)(col0 >> 8) * (3584 * 256);
  else if (col0 < 6144)
    wbase = Wkv + (size_t)((col0 - 4096) >> 8) * (3584 * 256);
  else
    wbase = Wkv + (size_t)(8 + ((col0 - 6144) >> 8)) * (3584 * 256);
  const int hb   = col0 & 255;
  const int arow = tid >> 2;
  const int ak   = (tid & 3) * 8;

  for (int kk = 0; kk < 3584; kk += 32) {
    // A tile 64x32 (f32 -> bf16)
    const float* ap = X + (size_t)(bm + arow) * 3584 + kk + ak;
    __builtin_prefetch(ap + 32, 0, 0);
    const float4 a0 = *(const float4*)(ap);
    const float4 a1 = *(const float4*)(ap + 4);
    __bf16* da = sA + arow * 34 + ak;
    da[0] = (__bf16)a0.x; da[1] = (__bf16)a0.y; da[2] = (__bf16)a0.z; da[3] = (__bf16)a0.w;
    da[4] = (__bf16)a1.x; da[5] = (__bf16)a1.y; da[6] = (__bf16)a1.z; da[7] = (__bf16)a1.w;
    // B tile 32x128 (f32 -> bf16, transposed into LDS)
    const float* bp = wbase + (size_t)(kk + krow) * 256 + hb;
    __builtin_prefetch(bp + 32 * 256, 0, 0);
#pragma unroll
    for (int j4 = 0; j4 < 4; ++j4) {
      const float4 bv = *(const float4*)(bp + j4 * 4);
      sB[(nb + j4 * 4 + 0) * 34 + krow] = (__bf16)bv.x;
      sB[(nb + j4 * 4 + 1) * 34 + krow] = (__bf16)bv.y;
      sB[(nb + j4 * 4 + 2) * 34 + krow] = (__bf16)bv.z;
      sB[(nb + j4 * 4 + 3) * 34 + krow] = (__bf16)bv.w;
    }
    __syncthreads();

    const bf16x16 af0 = frag_a_lds(sA + (wm + 0) * 34, 34, lane);
    const bf16x16 af1 = frag_a_lds(sA + (wm + 16) * 34, 34, lane);
    const bf16x16 bf0 = frag_b_lds(sB + (wn + 0) * 34, 34, lane);
    const bf16x16 bf1 = frag_b_lds(sB + (wn + 16) * 34, 34, lane);
    acc[0][0] = wmma_bf16(af0, bf0, acc[0][0]);
    acc[0][1] = wmma_bf16(af0, bf1, acc[0][1]);
    acc[1][0] = wmma_bf16(af1, bf0, acc[1][0]);
    acc[1][1] = wmma_bf16(af1, bf1, acc[1][1]);
    __syncthreads();
  }

  const int n  = lane & 15;
  const int mo = (lane >> 4) * 8;
#pragma unroll
  for (int i = 0; i < 2; ++i)
#pragma unroll
    for (int j = 0; j < 2; ++j)
#pragma unroll
      for (int g = 0; g < 8; ++g) {
        const int row = bm + wm + i * 16 + mo + g;
        const int col = bn + wn + j * 16 + n;
        Y[(size_t)row * 8192 + col] = (__bf16)acc[i][j][g];
      }
}

// ---------------------------------------------------------------------------
// Kernel 2: RMSNorm (+scale) then RoPE, in place on Y, for q and k heads.
// grid = (4096 rows, 24 head-vectors); block = 128 (one thread per rope pair)
// ---------------------------------------------------------------------------
__global__ __launch_bounds__(128) void k_norm_rope(__bf16* __restrict__ Y,
                                                   const float* __restrict__ qsc,
                                                   const float* __restrict__ ksc) {
  const int row = blockIdx.x;
  const int hv  = blockIdx.y;
  const int pos = row & 2047;  // t within sequence (row = b*2048 + t)
  const bool isq = hv < 16;
  __bf16* v = Y + (size_t)row * 8192 + (isq ? hv * 256 : 4096 + (hv - 16) * 256);
  const float* sc = isq ? qsc : ksc;

  const int i = threadIdx.x;  // 0..127
  float x1 = (float)v[i];
  float x2 = (float)v[i + 128];

  float ss = x1 * x1 + x2 * x2;
#pragma unroll
  for (int off = 16; off > 0; off >>= 1) ss += __shfl_xor(ss, off, 32);
  __shared__ float wsum[4];
  if ((i & 31) == 0) wsum[i >> 5] = ss;
  __syncthreads();
  const float tot = wsum[0] + wsum[1] + wsum[2] + wsum[3];
  const float r = rsqrtf(tot * (1.0f / 256.0f) + 1e-6f);

  x1 = x1 * r * sc[i];
  x2 = x2 * r * sc[i + 128];

  // timescale = 10000^(2i/256) -> angle = pos * exp(-i * ln(10000)/128)
  const float inv_ts = __expf(-(float)i * (9.210340371976184f / 128.0f));
  const float ang = (float)pos * inv_ts;
  float s, c;
  __sincosf(ang, &s, &c);
  v[i]       = (__bf16)(x1 * c - x2 * s);
  v[i + 128] = (__bf16)(x2 * c + x1 * s);
}

// ---------------------------------------------------------------------------
// Kernel 3: sliding-window flash attention with GQA K/V sharing.
// grid = (32 qtiles, 8 kv heads, 2 batch); block = 256 = 8 waves:
//   wave -> head = kv*2 + (wave&1), q-subtile = wave>>1 (16 rows each)
// Each wave: Q frags in regs, O accum 16x256 f32 in regs, online softmax.
// K tile (32x256 bf16, LDS rows padded to 264) staged by the Tensor Data
// Mover (tensor_load_to_lds with pad_interval=128dw/pad_amount=4dw), V tile
// staged manually transposed for the PV WMMA B-fragments.
// ---------------------------------------------------------------------------
#define ATT_WINDOW 1024

__global__ __launch_bounds__(256, 1) void k_attn(const __bf16* __restrict__ Y,
                                                 __bf16* __restrict__ Ao) {
  __shared__ __align__(16) __bf16 sK[32 * 264];   // [key][h], row stride 264
  __shared__ __bf16 sVt[256 * 34];                // [h][key]  (transposed V)
  __shared__ __bf16 sP[8][16 * 34];               // per-wave P tile [m][key]

  const int qt = blockIdx.x, kv = blockIdx.y, b = blockIdx.z;
  const int tid = threadIdx.x, lane = tid & 31, wave = tid >> 5;
  const int head = kv * 2 + (wave & 1);
  const int qsub = wave >> 1;               // 0..3
  const int qb   = qt * 64 + qsub * 16;     // first query row of this wave
  const size_t rowbase = (size_t)b * 2048;

  const __bf16* Qp = Y + (rowbase + qb) * 8192 + head * 256;
  const __bf16* Kp = Y + rowbase * 8192 + 4096 + kv * 256;
  const __bf16* Vp = Y + rowbase * 8192 + 6144 + kv * 256;

  // Q fragments: 8 chunks of 16x32 over head dim (held in registers)
  bf16x16 qf[8];
#pragma unroll
  for (int c = 0; c < 8; ++c) qf[c] = frag_a_glb(Qp + c * 32, 8192, lane);

  f32x8 oacc[16] = {};
  float rmax[8], rsum[8];
#pragma unroll
  for (int g = 0; g < 8; ++g) { rmax[g] = -1e30f; rsum[g] = 0.0f; }

  const int half = lane >> 4;
  const int nlan = lane & 15;
  int kstart = qt * 64 - (ATT_WINDOW - 1);
  if (kstart < 0) kstart = 0;
  kstart &= ~31;
  const int kend = qt * 64 + 64;  // exclusive

  for (int kb = kstart; kb < kend; kb += 32) {
#if USE_TDM
    // --- K tile via Tensor Data Mover: 32 rows x 512B, global row stride
    //     16KB, LDS row padded 512B -> 528B (pad 4 dwords every 128 dwords).
    if (wave == 0) {
      const unsigned lds_base = (unsigned)(size_t)(&sK[0]);
      const unsigned long long ga =
          (unsigned long long)(size_t)(Kp + (size_t)kb * 8192);
      u32x4 g0;
      g0.x = 1u;                                         // count=1, user mode
      g0.y = lds_base;                                   // lds_addr
      g0.z = (unsigned)(ga & 0xFFFFFFFFu);               // global_addr[31:0]
      g0.w = (unsigned)((ga >> 32) & 0x01FFFFFFu) | 0x80000000u; // [56:32]|type=2
      i32x8 g1;
      g1[0] = (1 << 16)     // data_size = 2 bytes
            | (1 << 20)     // pad_enable
            | (6 << 22)     // pad_interval: 128 dwords
            | (3 << 25);    // pad_amount: 4 dwords
      g1[1] = (int)((256u & 0xFFFFu) << 16);             // tensor_dim0[15:0]
      g1[2] = (int)(((256u >> 16) & 0xFFFFu) | ((32u & 0xFFFFu) << 16)); // dim0 hi | dim1 lo
      g1[3] = (int)(((32u >> 16) & 0xFFFFu) | (256u << 16)); // dim1 hi | tile_dim0=256
      g1[4] = (int)(32u);                                // tile_dim1=32, tile_dim2=0
      g1[5] = (int)(8192u);                              // tensor_dim0_stride lo
      g1[6] = 0;                                         // stride hi | dim1_stride lo
      g1[7] = 0;
      const i32x4 z4 = {0, 0, 0, 0};
      const i32x8 z8 = {0, 0, 0, 0, 0, 0, 0, 0};
      __builtin_amdgcn_tensor_load_to_lds(g0, g1, z4, z4, z8, 0);
      __builtin_amdgcn_s_wait_tensorcnt((short)0);
    }
    // V tile staged manually (needs transpose for PV B-fragments)
#pragma unroll
    for (int i = 0; i < 4; ++i) {
      const int c   = tid + 256 * i;   // 0..1023
      const int key = c >> 5;          // 0..31
      const int h   = (c & 31) * 8;    // 0..248
      const uint4 vvec = *(const uint4*)(Vp + (size_t)(kb + key) * 8192 + h);
      const __bf16* v8 = (const __bf16*)&vvec;
#pragma unroll
      for (int j = 0; j < 8; ++j) sVt[(h + j) * 34 + key] = v8[j];
    }
#else
    // fallback: cooperative staging of K (row-major) and V (transposed)
#pragma unroll
    for (int i = 0; i < 4; ++i) {
      const int c   = tid + 256 * i;
      const int key = c >> 5;
      const int h   = (c & 31) * 8;
      const uint4 kvec = *(const uint4*)(Kp + (size_t)(kb + key) * 8192 + h);
      *(uint4*)(sK + key * 264 + h) = kvec;
      const uint4 vvec = *(const uint4*)(Vp + (size_t)(kb + key) * 8192 + h);
      const __bf16* v8 = (const __bf16*)&vvec;
#pragma unroll
      for (int j = 0; j < 8; ++j) sVt[(h + j) * 34 + key] = v8[j];
    }
#endif
    __syncthreads();

    if (kb <= qb + 15 && kb + 31 >= qb - (ATT_WINDOW - 1)) {
      // S = Q * K^T  (two 16x16 tiles covering keys kb..kb+31)
      f32x8 s0 = {}, s1 = {};
#pragma unroll
      for (int c = 0; c < 8; ++c) {
        const bf16x16 b0 = frag_b_lds(sK + 0 * 264 + c * 32, 264, lane);
        const bf16x16 b1 = frag_b_lds(sK + 16 * 264 + c * 32, 264, lane);
        s0 = wmma_bf16(qf[c], b0, s0);
        s1 = wmma_bf16(qf[c], b1, s1);
      }
      // masked online softmax; C layout: row m = g + 8*half, col n = nlan
      float p0[8], p1[8], alpha[8];
#pragma unroll
      for (int g = 0; g < 8; ++g) {
        const int qpos = qb + g + half * 8;
        const int k0 = kb + nlan;
        const int k1 = kb + 16 + nlan;
        const bool v0 = (k0 <= qpos) && (k0 > qpos - ATT_WINDOW);
        const bool v1 = (k1 <= qpos) && (k1 > qpos - ATT_WINDOW);
        const float x0 = v0 ? s0[g] * 0.0625f : -1e30f;
        const float x1 = v1 ? s1[g] * 0.0625f : -1e30f;
        float mx = fmaxf(x0, x1);
#pragma unroll
        for (int m = 1; m < 16; m <<= 1) mx = fmaxf(mx, __shfl_xor(mx, m, 32));
        const float nm = fmaxf(rmax[g], mx);
        alpha[g] = __expf(rmax[g] - nm);
        rmax[g] = nm;
        p0[g] = v0 ? __expf(x0 - nm) : 0.0f;
        p1[g] = v1 ? __expf(x1 - nm) : 0.0f;
        float ps = p0[g] + p1[g];
#pragma unroll
        for (int m = 1; m < 16; m <<= 1) ps += __shfl_xor(ps, m, 32);
        rsum[g] = rsum[g] * alpha[g] + ps;
      }
      // rescale running output
#pragma unroll
      for (int c = 0; c < 16; ++c)
#pragma unroll
        for (int g = 0; g < 8; ++g) oacc[c][g] *= alpha[g];
      // C-layout -> A-layout for P via per-wave LDS (in-wave DS ordering)
      __bf16* pw = sP[wave];
#pragma unroll
      for (int g = 0; g < 8; ++g) {
        const int m = g + half * 8;
        pw[m * 34 + nlan]      = (__bf16)p0[g];
        pw[m * 34 + 16 + nlan] = (__bf16)p1[g];
      }
      const bf16x16 pf = frag_a_lds(pw, 34, lane);
      // O += P * V  (16 h-chunks of 16 columns)
#pragma unroll
      for (int c = 0; c < 16; ++c) {
        const bf16x16 vf = frag_b_lds(sVt + c * 16 * 34, 34, lane);
        oacc[c] = wmma_bf16(pf, vf, oacc[c]);
      }
    }
    __syncthreads();
  }

  // epilogue: divide by softmax sum, store bf16 attention output
#pragma unroll
  for (int g = 0; g < 8; ++g) rsum[g] = 1.0f / rsum[g];
#pragma unroll
  for (int c = 0; c < 16; ++c)
#pragma unroll
    for (int g = 0; g < 8; ++g) {
      const int row = qb + g + half * 8;
      const int col = head * 256 + c * 16 + nlan;
      Ao[(rowbase + row) * 4096 + col] = (__bf16)(oacc[c][g] * rsum[g]);
    }
}

// ---------------------------------------------------------------------------
// Kernel 4: output projection. A[4096][4096]bf16 x Wo[4096][3584]f32 -> f32
// ---------------------------------------------------------------------------
__global__ __launch_bounds__(256) void k_gemm_out(const __bf16* __restrict__ A,
                                                  const float* __restrict__ Wo,
                                                  float* __restrict__ Out) {
  __shared__ __bf16 sA[64 * 34];
  __shared__ __bf16 sB[128 * 34];  // transposed
  const int tid  = threadIdx.x;
  const int lane = tid & 31;
  const int wave = tid >> 5;
  const int bm = blockIdx.x * 64;
  const int bn = blockIdx.y * 128;
  const int wm = (wave >> 2) * 32;
  const int wn = (wave & 3) * 32;

  f32x8 acc[2][2] = {};

  const int krow = tid >> 3;
  const int nb   = (tid & 7) * 16;
  const int arow = tid >> 2;
  const int ak   = (tid & 3) * 8;

  for (int kk = 0; kk < 4096; kk += 32) {
    const __bf16* ap = A + (size_t)(bm + arow) * 4096 + kk + ak;
    __builtin_prefetch(ap + 32, 0, 0);
    const uint4 av = *(const uint4*)ap;
    const __bf16* a8 = (const __bf16*)&av;
    __bf16* da = sA + arow * 34 + ak;
#pragma unroll
    for (int i = 0; i < 8; ++i) da[i] = a8[i];

    const float* bp = Wo + (size_t)(kk + krow) * 3584 + bn + nb;
    __builtin_prefetch(bp + 32 * 3584, 0, 0);
#pragma unroll
    for (int j4 = 0; j4 < 4; ++j4) {
      const float4 bv = *(const float4*)(bp + j4 * 4);
      sB[(nb + j4 * 4 + 0) * 34 + krow] = (__bf16)bv.x;
      sB[(nb + j4 * 4 + 1) * 34 + krow] = (__bf16)bv.y;
      sB[(nb + j4 * 4 + 2) * 34 + krow] = (__bf16)bv.z;
      sB[(nb + j4 * 4 + 3) * 34 + krow] = (__bf16)bv.w;
    }
    __syncthreads();

    const bf16x16 af0 = frag_a_lds(sA + (wm + 0) * 34, 34, lane);
    const bf16x16 af1 = frag_a_lds(sA + (wm + 16) * 34, 34, lane);
    const bf16x16 bf0 = frag_b_lds(sB + (wn + 0) * 34, 34, lane);
    const bf16x16 bf1 = frag_b_lds(sB + (wn + 16) * 34, 34, lane);
    acc[0][0] = wmma_bf16(af0, bf0, acc[0][0]);
    acc[0][1] = wmma_bf16(af0, bf1, acc[0][1]);
    acc[1][0] = wmma_bf16(af1, bf0, acc[1][0]);
    acc[1][1] = wmma_bf16(af1, bf1, acc[1][1]);
    __syncthreads();
  }

  const int n  = lane & 15;
  const int mo = (lane >> 4) * 8;
#pragma unroll
  for (int i = 0; i < 2; ++i)
#pragma unroll
    for (int j = 0; j < 2; ++j)
#pragma unroll
      for (int g = 0; g < 8; ++g) {
        const int row = bm + wm + i * 16 + mo + g;
        const int col = bn + wn + j * 16 + n;
        Out[(size_t)row * 3584 + col] = acc[i][j][g];
      }
}

// ---------------------------------------------------------------------------
extern "C" void kernel_launch(void* const* d_in, const int* in_sizes, int n_in,
                              void* d_out, int out_size, void* d_ws, size_t ws_size,
                              hipStream_t stream) {
  (void)in_sizes; (void)n_in; (void)out_size; (void)ws_size;
  const float* x   = (const float*)d_in[0];
  const float* wq  = (const float*)d_in[1];
  const float* wkv = (const float*)d_in[2];
  const float* wo  = (const float*)d_in[3];
  const float* qsc = (const float*)d_in[4];
  const float* ksc = (const float*)d_in[5];
  float* out = (float*)d_out;

  __bf16* Y  = (__bf16*)d_ws;                                    // [4096][8192] bf16
  __bf16* Ao = (__bf16*)((char*)d_ws + (size_t)4096 * 8192 * 2); // [4096][4096] bf16

  k_gemm_qkv<<<dim3(64, 64), 256, 0, stream>>>(x, wq, wkv, Y);
  k_norm_rope<<<dim3(4096, 24), 128, 0, stream>>>(Y, qsc, ksc);
  k_attn<<<dim3(32, 8, 2), 256, 0, stream>>>(Y, Ao);
  k_gemm_out<<<dim3(64, 28), 256, 0, stream>>>(Ao, wo, out);
}
